// S4Block_85779086835963
// MI455X (gfx1250) — compile-verified, matrix operand order checked by reading, not run on hardware
//
#include <hip/hip_runtime.h>
#include <hip/hip_bf16.h>
#include <math.h>

#define BB 4
#define HH 1024
#define LL 4096
#define NMODES 32
#define QQ 64        // chunk length
#define NCH 64       // LL/QQ chunks

typedef __attribute__((ext_vector_type(16))) __bf16 v16bf;
typedef __attribute__((ext_vector_type(8)))  float  v8f;
typedef __attribute__((ext_vector_type(4)))  unsigned int v4u;
typedef __attribute__((ext_vector_type(8)))  int v8i;
typedef __attribute__((ext_vector_type(4)))  int v4i;

union Frag16 { v16bf v; uint4 u[2]; };

__device__ inline float2 cmul(float2 a, float2 b) {
    return make_float2(a.x * b.x - a.y * b.y, a.x * b.y + a.y * b.x);
}

// ---------------------------------------------------------------------------
// Kernel 1: per-(h,n) S4D coefficients. One wave per h, lane = mode n.
//   w    = exp(dt*A)            (per-step complex decay)
//   wQ   = w^Q                  (per-chunk decay)
//   Ceff = (C_re+iC_im)*expm1(dtA)/A
//   kch[h][t] = 2*Re(sum_n Ceff_n w_n^t), t<Q   (intra-chunk Toeplitz taps)
// ---------------------------------------------------------------------------
__global__ void s4d_coeffs(const float* __restrict__ log_dt,
                           const float* __restrict__ A_real_log,
                           const float* __restrict__ A_imag,
                           const float* __restrict__ C_re,
                           const float* __restrict__ C_im,
                           float2* __restrict__ w_out,
                           float2* __restrict__ wq_out,
                           float2* __restrict__ ceff_out,
                           float*  __restrict__ kch_out)
{
    int h = blockIdx.x;
    int n = threadIdx.x;
    int idx = h * NMODES + n;

    float dt  = expf(log_dt[h]);
    float ar  = -expf(A_real_log[idx]);
    float ai  = A_imag[idx];
    float dre = ar * dt, dim = ai * dt;

    float e  = expf(dre);
    float2 w = make_float2(e * cosf(dim), e * sinf(dim));
    float2 em1 = make_float2(w.x - 1.0f, w.y);            // expm1(dtA)
    float den = ar * ar + ai * ai;
    float2 eoa = make_float2((em1.x * ar + em1.y * ai) / den,
                             (em1.y * ar - em1.x * ai) / den);   // expm1(dtA)/A
    float2 c = make_float2(C_re[idx], C_im[idx]);
    float2 ceff = cmul(c, eoa);

    float eq = expf(dre * (float)QQ);
    float aq = dim * (float)QQ;
    float2 wq = make_float2(eq * cosf(aq), eq * sinf(aq));

    w_out[idx]    = w;
    wq_out[idx]   = wq;
    ceff_out[idx] = ceff;

    // kernel taps: z = Ceff * w^t, reduce Re over the 32 lanes (wave32)
    float2 z = ceff;
    for (int t = 0; t < QQ; ++t) {
        float s = z.x;
        for (int m = 16; m; m >>= 1) s += __shfl_xor(s, m, 32);
        if (n == 0) kch_out[h * QQ + t] = 2.0f * s;
        z = cmul(z, w);
    }
}

// ---------------------------------------------------------------------------
// Kernel 2b: W_out (H x 2H fp32) -> Wt (2H x H bf16), k-contiguous rows so the
// GEMM B-fragment is two 16B loads per lane.
// ---------------------------------------------------------------------------
__global__ void w_transpose(const float* __restrict__ W,
                            __hip_bfloat16* __restrict__ Wt)
{
    __shared__ float tile[32][33];
    int n0 = blockIdx.x * 32, k0 = blockIdx.y * 32;
    int c = threadIdx.x & 31, r0 = threadIdx.x >> 5;
    #pragma unroll
    for (int j = 0; j < 4; ++j) {
        int r = r0 + j * 8;
        tile[r][c] = W[(size_t)(k0 + r) * (2 * HH) + n0 + c];
    }
    __syncthreads();
    #pragma unroll
    for (int j = 0; j < 4; ++j) {
        int r = r0 + j * 8;
        Wt[(size_t)(n0 + r) * HH + k0 + c] = __float2bfloat16(tile[c][r]);
    }
}

// ---------------------------------------------------------------------------
// Kernel 2: chunked SSM scan, one workgroup per (b,h), 8 waves, lane = mode.
// x row staged into LDS by the Tensor Data Mover (TDM), D# built inline.
// Phases: (1) per-chunk Horner projections P[c]  (2) 64-step inter-chunk scan
// + C*w^(i+1) table  (3) state-out + intra-chunk Toeplitz + D skip + GELU.
// ---------------------------------------------------------------------------
__global__ __launch_bounds__(256) void s4d_scan(
    const float*  __restrict__ x,
    const float2* __restrict__ wv,
    const float2* __restrict__ wqv,
    const float2* __restrict__ ceffv,
    const float*  __restrict__ kch,
    const float*  __restrict__ Dv,
    __hip_bfloat16* __restrict__ ybf)
{
    __shared__ float  lds_x[LL];                   // 16KB x row
    __shared__ float2 lds_PS[NCH * NMODES];        // P[c], overwritten by state-prefix
    __shared__ float2 lds_Cw[NMODES * (QQ + 1)];   // [n][i] (padded): (2Re, -2Im) of Ceff*w^(i+1)
    __shared__ float  lds_k[QQ];

    int bh = blockIdx.x;
    int b = bh / HH, h = bh % HH;
    int tid = threadIdx.x, wave = tid >> 5, lane = tid & 31;

    const float* xrow = x + ((size_t)b * HH + h) * LL;

    if (tid == 0) {
        // --- Tensor Data Mover: 1-D tile, 4096 fp32 -> LDS ---
        unsigned lds_off = (unsigned)(unsigned long long)
            (__attribute__((address_space(3))) void*)lds_x;
        unsigned long long ga = (unsigned long long)(const void*)xrow;
        v4u g0;
        g0[0] = 1u;                                          // count=1, no gather
        g0[1] = lds_off;                                     // lds_addr (bytes)
        g0[2] = (unsigned)(ga & 0xFFFFFFFFull);              // global_addr[31:0]
        g0[3] = (unsigned)((ga >> 32) & 0x1FFFFFFull) | (2u << 30); // addr[56:32] | type=2
        v8i g1;
        g1[0] = 0x00020000;                                  // data_size=4B, mask=0
        g1[1] = (int)(LL << 16);                             // tensor_dim0[15:0]<<16
        g1[2] = (int)((LL >> 16) | (1 << 16));               // dim0[31:16] | tensor_dim1=1
        g1[3] = (int)(LL << 16);                             // tile_dim0=4096
        g1[4] = 1;                                           // tile_dim1=1, tile_dim2=0
        g1[5] = LL;                                          // tensor_dim0_stride
        g1[6] = 0; g1[7] = 0;
        v4i g2 = {0, 0, 0, 0};
        v4i g3 = {0, 0, 0, 0};
        v8i g4 = {0, 0, 0, 0, 0, 0, 0, 0};                   // unused group (6-arg form)
        __builtin_amdgcn_tensor_load_to_lds(g0, g1, g2, g3, g4, 0);
        __builtin_amdgcn_s_wait_tensorcnt(0);
    }
    if (tid >= 64 && tid < 128) lds_k[tid - 64] = kch[h * QQ + (tid - 64)];

    float2 w  = wv[h * NMODES + lane];
    float2 wq = wqv[h * NMODES + lane];
    float2 ce = ceffv[h * NMODES + lane];
    float  Dh = Dv[h];
    __syncthreads();

    // Phase 1: P[c] = sum_j w^(Q-1-j) x[cQ+j] via Horner, 8 chunks per wave
    for (int c = wave; c < NCH; c += 8) {
        float2 p = make_float2(0.f, 0.f);
        const float* xc = &lds_x[c * QQ];
        for (int j = 0; j < QQ; ++j) {
            float xv = xc[j];
            p = make_float2(p.x * w.x - p.y * w.y + xv,
                            p.x * w.y + p.y * w.x);
        }
        lds_PS[c * NMODES + lane] = p;
    }
    __syncthreads();

    // Phase 2: wave0 = inter-chunk scan (in place: P[c] -> state before chunk c)
    //          wave1 = Cw table: (2Re, -2Im) of Ceff * w^(i+1)
    if (wave == 0) {
        float2 s = make_float2(0.f, 0.f);
        for (int c = 0; c < NCH; ++c) {
            float2 pc = lds_PS[c * NMODES + lane];
            lds_PS[c * NMODES + lane] = s;
            s = make_float2(s.x * wq.x - s.y * wq.y + pc.x,
                            s.x * wq.y + s.y * wq.x + pc.y);
        }
    } else if (wave == 1) {
        float2 z = cmul(ce, w);
        for (int i = 0; i < QQ; ++i) {
            lds_Cw[lane * (QQ + 1) + i] = make_float2(2.f * z.x, -2.f * z.y);
            z = cmul(z, w);
        }
    }
    __syncthreads();

    // Phase 3: y[cQ+i] = state-out + intra-chunk Toeplitz + D*x, then exact GELU
    for (int c = wave; c < NCH; c += 8) {
        const float2* sc = &lds_PS[c * NMODES];
        const float*  xc = &lds_x[c * QQ];
        #pragma unroll
        for (int rep = 0; rep < 2; ++rep) {
            int i = lane + rep * 32;
            float ys = 0.f;
            for (int n2 = 0; n2 < NMODES; ++n2) {
                float2 cw = lds_Cw[n2 * (QQ + 1) + i];
                float2 s  = sc[n2];
                ys += cw.x * s.x + cw.y * s.y;
            }
            float yi = 0.f;
            for (int j = 0; j <= i; ++j) yi += lds_k[i - j] * xc[j];
            float y = ys + yi + Dh * xc[i];
            float g = 0.5f * y * (1.f + erff(y * 0.70710678118654752f));
            ybf[((size_t)b * HH + h) * LL + c * QQ + i] = __float2bfloat16(g);
        }
    }
}

// ---------------------------------------------------------------------------
// Kernel 2c: ybf (B,H,L) -> yT (B*L, H) so GEMM A-fragments are k-contiguous.
// ---------------------------------------------------------------------------
__global__ void y_transpose(const __hip_bfloat16* __restrict__ ybf,
                            __hip_bfloat16* __restrict__ yT)
{
    __shared__ unsigned short tile[64][65];
    int b = blockIdx.z, h0 = blockIdx.y * 64, l0 = blockIdx.x * 64;
    int t = threadIdx.x;
    int hr = t >> 2, lc0 = (t & 3) * 16;
    const unsigned short* src = (const unsigned short*)ybf
        + ((size_t)b * HH + h0 + hr) * LL + l0 + lc0;
    #pragma unroll
    for (int j = 0; j < 16; ++j) tile[hr][lc0 + j] = src[j];
    __syncthreads();
    int lr = t >> 2, hc0 = (t & 3) * 16;
    unsigned short* dst = (unsigned short*)yT
        + ((size_t)b * LL + l0 + lr) * HH + h0 + hc0;
    #pragma unroll
    for (int j = 0; j < 16; ++j) dst[j] = tile[hc0 + j][lr];
}

// ---------------------------------------------------------------------------
// Kernel 3: Z = yT @ Wt^T (+bias), GLU over halves, write out (B,H,L) fp32.
// 8 waves = 4(m) x 2(n); wave tile 32m x 32n x {a,g} halves -> 8 WMMA / k-step.
// Fragments loaded straight from global (L2-resident), 2 x 16B per lane per
// fragment matching the documented 16-bit A/B VGPR layouts.
// ---------------------------------------------------------------------------
__global__ __launch_bounds__(256) void glu_gemm(
    const __hip_bfloat16* __restrict__ yT,
    const __hip_bfloat16* __restrict__ Wt,
    const float* __restrict__ bias,
    float* __restrict__ out)
{
    int tid = threadIdx.x, wave = tid >> 5, lane = tid & 31;
    int mw = wave & 3, nw = wave >> 2;
    int m_base = blockIdx.y * 128 + mw * 32;
    int n_base = blockIdx.x * 64 + nw * 32;
    int row = lane & 15;
    int g   = (lane >> 4) * 8;           // k-group select per lane half
    int b = m_base / LL;
    int l_base = m_base % LL;

    v8f acc[2][2][2] = {};
    const unsigned short* A  = (const unsigned short*)yT;
    const unsigned short* Bp = (const unsigned short*)Wt;

    for (int k0 = 0; k0 < HH; k0 += 32) {
        Frag16 af[2];
        Frag16 bfr[2][2];
        #pragma unroll
        for (int mt = 0; mt < 2; ++mt) {
            const unsigned short* ap =
                A + ((size_t)(m_base + mt * 16 + row)) * HH + k0 + g;
            af[mt].u[0] = *(const uint4*)ap;
            af[mt].u[1] = *(const uint4*)(ap + 16);
            __builtin_prefetch(ap + 32, 0, 0);     // next k-block
        }
        #pragma unroll
        for (int nt = 0; nt < 2; ++nt)
            #pragma unroll
            for (int hf = 0; hf < 2; ++hf) {
                const unsigned short* bp =
                    Bp + ((size_t)(n_base + nt * 16 + row + hf * HH)) * HH + k0 + g;
                bfr[nt][hf].u[0] = *(const uint4*)bp;
                bfr[nt][hf].u[1] = *(const uint4*)(bp + 16);
            }
        #pragma unroll
        for (int mt = 0; mt < 2; ++mt)
            #pragma unroll
            for (int nt = 0; nt < 2; ++nt)
                #pragma unroll
                for (int hf = 0; hf < 2; ++hf)
                    acc[mt][nt][hf] = __builtin_amdgcn_wmma_f32_16x16x32_bf16(
                        false, af[mt].v, false, bfr[nt][hf].v,
                        (short)0, acc[mt][nt][hf], false, false);
    }

    // Epilogue: bias + GLU, 2x float4 (32B contiguous along l) per lane per tile
    #pragma unroll
    for (int mt = 0; mt < 2; ++mt)
        #pragma unroll
        for (int nt = 0; nt < 2; ++nt) {
            int n = n_base + nt * 16 + row;
            float b1 = bias[n], b2 = bias[n + HH];
            float o[8];
            #pragma unroll
            for (int v = 0; v < 8; ++v) {
                float z1 = acc[mt][nt][0][v] + b1;
                float z2 = acc[mt][nt][1][v] + b2;
                o[v] = z1 / (1.f + expf(-z2));
            }
            int ml = mt * 16 + g;
            float* dst = out + ((size_t)b * HH + n) * LL + l_base + ml;
            *(float4*)dst       = make_float4(o[0], o[1], o[2], o[3]);
            *(float4*)(dst + 4) = make_float4(o[4], o[5], o[6], o[7]);
        }
}

// ---------------------------------------------------------------------------
extern "C" void kernel_launch(void* const* d_in, const int* in_sizes, int n_in,
                              void* d_out, int out_size, void* d_ws, size_t ws_size,
                              hipStream_t stream)
{
    (void)in_sizes; (void)n_in; (void)out_size; (void)ws_size;
    const float* x          = (const float*)d_in[0];
    const float* log_dt     = (const float*)d_in[1];
    const float* A_real_log = (const float*)d_in[2];
    const float* A_imag     = (const float*)d_in[3];
    const float* C_re       = (const float*)d_in[4];
    const float* C_im       = (const float*)d_in[5];
    const float* Dv         = (const float*)d_in[6];
    const float* W_out      = (const float*)d_in[7];
    const float* b_out      = (const float*)d_in[8];

    char* ws = (char*)d_ws;
    float2* wv   = (float2*)(ws + 0x000000);          // 256KB
    float2* wqv  = (float2*)(ws + 0x040000);          // 256KB
    float2* ceff = (float2*)(ws + 0x080000);          // 256KB
    float*  kch  = (float*) (ws + 0x0C0000);          // 256KB
    __hip_bfloat16* Wt  = (__hip_bfloat16*)(ws + 0x100000);   // 4MB
    __hip_bfloat16* ybf = (__hip_bfloat16*)(ws + 0x500000);   // 32MB
    __hip_bfloat16* yT  = (__hip_bfloat16*)(ws + 0x2500000);  // 32MB

    s4d_coeffs<<<HH, NMODES, 0, stream>>>(log_dt, A_real_log, A_imag, C_re, C_im,
                                          wv, wqv, ceff, kch);
    w_transpose<<<dim3(2 * HH / 32, HH / 32), 256, 0, stream>>>(W_out, Wt);
    s4d_scan<<<BB * HH, 256, 0, stream>>>(x, wv, wqv, ceff, kch, Dv, ybf);
    y_transpose<<<dim3(LL / 64, HH / 64, BB), 256, 0, stream>>>(ybf, yT);
    glu_gemm<<<dim3(HH / 64, (BB * LL) / 128), 256, 0, stream>>>(yT, Wt, b_out,
                                                                 (float*)d_out);
}
